// DINForDurationPrediction_78099685310539
// MI455X (gfx1250) — compile-verified, matrix operand order checked by reading, not run on hardware
//
#include <hip/hip_runtime.h>
#include <math.h>

// ---------------------------------------------------------------------------
// Types for CDNA5 WMMA
// ---------------------------------------------------------------------------
typedef __attribute__((ext_vector_type(16))) _Float16 v16h;
typedef __attribute__((ext_vector_type(8)))  _Float16 v8h;
typedef __attribute__((ext_vector_type(4)))  _Float16 v4h;
typedef __attribute__((ext_vector_type(8)))  float    v8f;

#define BM 128
#define BN 64
#define BK 32
#define LDA 40   // padded LDS row stride in halves (32 + 8) -> conflict-free b128 reads

// ---------------------------------------------------------------------------
// CDNA5 async global->LDS copy (ASYNCcnt-tracked) via inline asm.
//   GLOBAL_LOAD_ASYNC_TO_LDS_B128: VDST = LDS byte address (single VGPR),
//   VADDR = 64-bit global address. LDS offset = low 32 bits of generic ptr.
// ---------------------------------------------------------------------------
#define USE_ASYNC 1

#if USE_ASYNC
__device__ __forceinline__ void async_b128(const void* g, void* l) {
  unsigned lds_addr = (unsigned)(unsigned long long)l;
  unsigned long long ga = (unsigned long long)g;
  asm volatile("global_load_async_to_lds_b128 %0, %1, off"
               :: "v"(lds_addr), "v"(ga) : "memory");
}
__device__ __forceinline__ void wait_async() {
  asm volatile("s_wait_asynccnt 0x0" ::: "memory");
}
#define HAVE_ASYNC_LDS 1
#else
#define HAVE_ASYNC_LDS 0
__device__ __forceinline__ void wait_async() {}
#endif

// ---------------------------------------------------------------------------
// Weight prep: fp32 [K,N] row-major -> f16 [N,K] (transposed) for WMMA B path
// ---------------------------------------------------------------------------
__global__ void k_transpose_h(const float* __restrict__ src, _Float16* __restrict__ dst,
                              int K, int N) {
  int total = K * N;
  for (int idx = blockIdx.x * blockDim.x + threadIdx.x; idx < total;
       idx += gridDim.x * blockDim.x) {
    int k = idx / N, n = idx - k * N;
    dst[(long)n * K + k] = (_Float16)src[idx];
  }
}

// ---------------------------------------------------------------------------
// Tiled WMMA GEMM with double-buffered LDS + software pipelining:
//   C[M,N] = A[M,K] @ W[K,N] + bias
//   A: fp32 (global->VGPR->cvt->LDS, loads overlapped with WMMA)
//      or f16 (GLOBAL_LOAD_ASYNC_TO_LDS_B128 straight into LDS)
//   Wt: pre-transposed f16 [N,K], staged via async-to-LDS.
// Requires M%128==0, N%64==0, K%32==0 (true for all GEMMs here).
// Block = 256 threads = 8 wave32 waves; wave (wy,wx) owns a 32x32 patch
// (2x2 V_WMMA_F32_16X16X32_F16 tiles).
// ---------------------------------------------------------------------------
template <typename AT>
__global__ __launch_bounds__(256)
void k_wmma_gemm(const AT* __restrict__ A, const _Float16* __restrict__ Wt,
                 const float* __restrict__ bias,
                 float* __restrict__ Cf, _Float16* __restrict__ Ch,
                 int M, int N, int K, int ldc) {
  __shared__ _Float16 sA[2][BM * LDA];
  __shared__ _Float16 sB[2][BN * LDA];

  const int tid  = threadIdx.x;
  const int lane = tid & 31;
  const int wid  = tid >> 5;
  const int wy   = wid & 3;   // 4 waves along M
  const int wx   = wid >> 2;  // 2 waves along N
  const long gm0 = (long)blockIdx.y * BM;
  const long gn0 = (long)blockIdx.x * BN;

  v8f acc[2][2];
#pragma unroll
  for (int i = 0; i < 2; i++)
#pragma unroll
    for (int j = 0; j < 2; j++)
#pragma unroll
      for (int e = 0; e < 8; e++) acc[i][j][e] = 0.0f;

  // ---- staging state (register-held between issue and finish phases) ----
  float4 va[4];     // f32-A path
#if !HAVE_ASYNC_LDS
  v8h vah[2];       // f16-A fallback
  v8h vb;           // B fallback
#endif

  // issue phase: put global loads in flight (or fire async-to-LDS copies)
  auto issue_stage = [&](int t, int nxt) {
    const int k0 = t * BK;
    if constexpr (__is_same(AT, float)) {
#pragma unroll
      for (int c = 0; c < 4; c++) {
        int r = (tid >> 3) + c * 32;
        va[c] = *(const float4*)(A + (gm0 + r) * (long)K + k0 + (tid & 7) * 4);
      }
    } else {
#pragma unroll
      for (int c = 0; c < 2; c++) {
        int r = (tid >> 2) + c * 64;
#if HAVE_ASYNC_LDS
        async_b128(A + (gm0 + r) * (long)K + k0 + (tid & 3) * 8,
                   &sA[nxt][r * LDA + (tid & 3) * 8]);
#else
        vah[c] = *(const v8h*)(A + (gm0 + r) * (long)K + k0 + (tid & 3) * 8);
#endif
      }
    }
    {
      int r = tid >> 2, c8 = tid & 3;
#if HAVE_ASYNC_LDS
      async_b128(Wt + (gn0 + r) * (long)K + k0 + c8 * 8,
                 &sB[nxt][r * LDA + c8 * 8]);
#else
      vb = *(const v8h*)(Wt + (gn0 + r) * (long)K + k0 + c8 * 8);
#endif
    }
    (void)nxt;
  };

  // finish phase: convert/store register-staged data into LDS
  auto finish_stage = [&](int nxt) {
    if constexpr (__is_same(AT, float)) {
#pragma unroll
      for (int c = 0; c < 4; c++) {
        int r = (tid >> 3) + c * 32;
        v4h h;
        h[0] = (_Float16)va[c].x; h[1] = (_Float16)va[c].y;
        h[2] = (_Float16)va[c].z; h[3] = (_Float16)va[c].w;
        *(v4h*)(&sA[nxt][r * LDA + (tid & 7) * 4]) = h;
      }
    }
#if !HAVE_ASYNC_LDS
    if constexpr (!__is_same(AT, float)) {
#pragma unroll
      for (int c = 0; c < 2; c++) {
        int r = (tid >> 2) + c * 64;
        *(v8h*)(&sA[nxt][r * LDA + (tid & 3) * 8]) = vah[c];
      }
    }
    { int r = tid >> 2, c8 = tid & 3;
      *(v8h*)(&sB[nxt][r * LDA + c8 * 8]) = vb; }
#endif
    (void)nxt;
  };

  const int nt = K / BK;
  // prologue: stage tile 0 into buffer 0
  issue_stage(0, 0);
  finish_stage(0);
  wait_async();
  __syncthreads();

  for (int t = 0; t < nt; ++t) {
    const int  cur  = t & 1;
    const int  nxt  = cur ^ 1;
    const bool more = (t + 1) < nt;

    // put next tile's global traffic in flight before doing any math
    if (more) issue_stage(t + 1, nxt);

    // ---- compute current tile: fragments per documented 16x32 f16 layouts ----
    const int grp  = (lane >> 4) << 3;   // lanes 16-31 handle K+8 / K+24 halves
    const int mrow = lane & 15;
    v16h a[2], b[2];
#pragma unroll
    for (int i = 0; i < 2; i++) {
      const _Float16* p = &sA[cur][(wy * 32 + i * 16 + mrow) * LDA + grp];
      v8h lo = *(const v8h*)p;
      v8h hi = *(const v8h*)(p + 16);
#pragma unroll
      for (int e = 0; e < 8; e++) { a[i][e] = lo[e]; a[i][8 + e] = hi[e]; }
    }
#pragma unroll
    for (int j = 0; j < 2; j++) {
      const _Float16* p = &sB[cur][(wx * 32 + j * 16 + mrow) * LDA + grp];
      v8h lo = *(const v8h*)p;
      v8h hi = *(const v8h*)(p + 16);
#pragma unroll
      for (int e = 0; e < 8; e++) { b[j][e] = lo[e]; b[j][8 + e] = hi[e]; }
    }
#pragma unroll
    for (int i = 0; i < 2; i++)
#pragma unroll
      for (int j = 0; j < 2; j++)
        acc[i][j] = __builtin_amdgcn_wmma_f32_16x16x32_f16(
            false, a[i], false, b[j], (short)0, acc[i][j], false, false);

    if (more) {
      finish_stage(nxt);  // loadcnt wait + cvt + ds_store happen after WMMAs
      wait_async();
      __syncthreads();    // buf[nxt] ready; all waves done reading buf[cur]
    }
  }

  // ---- epilogue: C 16x16 layout (lane n, VGPR r -> M = r + 8*(lane>=16)) ----
#pragma unroll
  for (int i = 0; i < 2; i++) {
#pragma unroll
    for (int j = 0; j < 2; j++) {
      int  gn = (int)gn0 + wx * 32 + j * 16 + (lane & 15);
      float bv = bias ? bias[gn] : 0.0f;
      long gmB = gm0 + wy * 32 + i * 16 + ((lane >> 4) << 3);
#pragma unroll
      for (int r = 0; r < 8; r++) {
        float v = acc[i][j][r] + bv;
        if (Cf) Cf[(gmB + r) * (long)ldc + gn] = v;
        if (Ch) Ch[(gmB + r) * (long)ldc + gn] = (_Float16)v;
      }
    }
  }
}

// ---------------------------------------------------------------------------
// Attention (simplified: score rows identical across queries) + meancat build.
// One wave32 per sample; lane owns 8 of 256 channels.
// ---------------------------------------------------------------------------
__global__ __launch_bounds__(256)
void k_attention(const float* __restrict__ q, const float* __restrict__ latest,
                 const _Float16* __restrict__ kv,   // [B*20, 512] keys|vals
                 const int* __restrict__ lengths, const float* __restrict__ ts,
                 _Float16* __restrict__ xh, _Float16* __restrict__ meancat) {
  const int lane = threadIdx.x & 31;
  const long b = (long)blockIdx.x * 8 + (threadIdx.x >> 5);

  float qr[8], mk[8];
  const float* qp = q + b * 256 + lane * 8;
#pragma unroll
  for (int i = 0; i < 8; i++) { qr[i] = qp[i]; mk[i] = 0.0f; }

  const int len = lengths[b];
  float sc[20];
  for (int k = 0; k < 20; k++) {
    v8h kh = *(const v8h*)(kv + (b * 20 + k) * 512 + lane * 8);
    float p = 0.0f;
#pragma unroll
    for (int i = 0; i < 8; i++) {
      float kf = (float)kh[i];
      p += qr[i] * kf;
      mk[i] += kf;
    }
#pragma unroll
    for (int m = 16; m >= 1; m >>= 1) p += __shfl_xor(p, m, 32);
    float tsv = ts[b * 20 + k];
    sc[k] = (k < len) ? (p * 0.125f) * __expf(-0.01f * tsv) : -1e30f;
  }

  float mx = sc[0];
  for (int k = 1; k < 20; k++) mx = fmaxf(mx, sc[k]);
  float w[20], s = 0.0f;
  for (int k = 0; k < 20; k++) { w[k] = __expf(sc[k] - mx); s += w[k]; }
  const float inv = 1.0f / s;

  float att[8];
#pragma unroll
  for (int i = 0; i < 8; i++) att[i] = 0.0f;
  for (int k = 0; k < 20; k++) {
    v8h vh = *(const v8h*)(kv + (b * 20 + k) * 512 + 256 + lane * 8);
    float wk = w[k] * inv;
#pragma unroll
    for (int i = 0; i < 8; i++) att[i] += wk * (float)vh[i];
  }

  const float* lp = latest + b * 256 + lane * 8;
  _Float16* xo = xh + b * 896 + 384 + lane * 8;
  _Float16* mc = meancat + b * 1024 + lane * 8;
#pragma unroll
  for (int i = 0; i < 8; i++) {
    xo[i] = (_Float16)(att[i] * 20.0f + lp[i]);   // attended.sum + latest
    float qv = qr[i], mv = mk[i] * (1.0f / 20.0f);
    mc[i]       = (_Float16)qv;
    mc[256 + i] = (_Float16)mv;
    mc[512 + i] = (_Float16)(qv - mv);
    mc[768 + i] = (_Float16)(qv * mv);
  }
}

// ---------------------------------------------------------------------------
// Numeric encoder + categorical embeddings + latest -> x[:,0:384)
// One wave32 per sample.
// ---------------------------------------------------------------------------
__global__ __launch_bounds__(256)
void k_encoders(const float* __restrict__ nf, const int* __restrict__ c0,
                const int* __restrict__ c1, const int* __restrict__ c2,
                const float* __restrict__ e0, const float* __restrict__ e1,
                const float* __restrict__ e2,
                const float* __restrict__ w1, const float* __restrict__ b1,
                const float* __restrict__ w2, const float* __restrict__ b2,
                const float* __restrict__ latest, _Float16* __restrict__ xh) {
  const int lane = threadIdx.x & 31;
  const long b = (long)blockIdx.x * 8 + (threadIdx.x >> 5);
  _Float16* x = xh + b * 896;

  // layer1: 64 -> 32, lane owns one output
  float h1 = b1[lane];
  for (int i = 0; i < 64; i++) h1 += nf[b * 64 + i] * w1[i * 32 + lane];
  h1 = fmaxf(h1, 0.0f);
  // layer2: 32 -> 16 via shuffles (uniform loop, no divergence on shfl)
  float h2 = (lane < 16) ? b2[lane] : 0.0f;
  for (int j = 0; j < 32; j++) {
    float hv = __shfl(h1, j, 32);
    if (lane < 16) h2 += hv * w2[j * 16 + lane];
  }
  if (lane < 16) x[lane] = (_Float16)fmaxf(h2, 0.0f);

  // categorical embeddings: [emb0(32) | emb1(16) | emb2(64)] -> x[16:128)
  x[16 + lane] = (_Float16)e0[(long)c0[b] * 32 + lane];
  if (lane < 16) x[48 + lane] = (_Float16)e1[(long)c1[b] * 16 + lane];
  x[64 + lane] = (_Float16)e2[(long)c2[b] * 64 + lane];
  x[96 + lane] = (_Float16)e2[(long)c2[b] * 64 + 32 + lane];

  // latest -> x[128:384)
#pragma unroll
  for (int i = 0; i < 8; i++)
    x[128 + lane * 8 + i] = (_Float16)latest[b * 256 + lane * 8 + i];
}

// ---------------------------------------------------------------------------
// Fused LayerNorm + ReLU, one wave32 per row (D = 128 or 64).
// ---------------------------------------------------------------------------
__global__ __launch_bounds__(256)
void k_ln_relu(const float* __restrict__ in, const float* __restrict__ g,
               const float* __restrict__ bb, _Float16* __restrict__ outh,
               float* __restrict__ outf, int D) {
  const int lane = threadIdx.x & 31;
  const long r = (long)blockIdx.x * 8 + (threadIdx.x >> 5);
  const float* p = in + r * D;
  const int per = D >> 5;  // <= 4
  float v[4], s = 0.0f, s2 = 0.0f;
  for (int j = 0; j < per; j++) {
    v[j] = p[lane + j * 32];
    s += v[j];
    s2 += v[j] * v[j];
  }
#pragma unroll
  for (int m = 16; m >= 1; m >>= 1) {
    s += __shfl_xor(s, m, 32);
    s2 += __shfl_xor(s2, m, 32);
  }
  const float mean = s / (float)D;
  const float var = s2 / (float)D - mean * mean;
  const float rstd = rsqrtf(var + 1e-5f);
  for (int j = 0; j < per; j++) {
    int c = lane + j * 32;
    float o = (v[j] - mean) * rstd * g[c] + bb[c];
    o = fmaxf(o, 0.0f);
    if (outh) outh[r * D + c] = (_Float16)o;
    if (outf) outf[r * D + c] = o;
  }
}

// ---------------------------------------------------------------------------
// Final 64 -> 1 GEMV, one wave32 per sample.
// ---------------------------------------------------------------------------
__global__ __launch_bounds__(256)
void k_fc4(const float* __restrict__ h, const float* __restrict__ w,
           const float* __restrict__ bz, float* __restrict__ out) {
  const int lane = threadIdx.x & 31;
  const long b = (long)blockIdx.x * 8 + (threadIdx.x >> 5);
  float p = h[b * 64 + lane] * w[lane] + h[b * 64 + 32 + lane] * w[32 + lane];
#pragma unroll
  for (int m = 16; m >= 1; m >>= 1) p += __shfl_xor(p, m, 32);
  if (lane == 0) out[b] = p + bz[0];
}

// ---------------------------------------------------------------------------
// Host launcher
// ---------------------------------------------------------------------------
extern "C" void kernel_launch(void* const* d_in, const int* in_sizes, int n_in,
                              void* d_out, int out_size, void* d_ws, size_t ws_size,
                              hipStream_t stream) {
  (void)in_sizes; (void)n_in; (void)out_size; (void)ws_size;
  const int B = 8192, S = 20, TXT = 768, EMB = 256;
  const long BS = (long)B * S;

  // ---- inputs (setup_inputs order) ----
  const float* num_features = (const float*)d_in[0];
  const int*   cat0 = (const int*)d_in[1];
  const int*   cat1 = (const int*)d_in[2];
  const int*   cat2 = (const int*)d_in[3];
  const float* past_text = (const float*)d_in[4];
  const int*   past_len  = (const int*)d_in[5];
  const float* cand      = (const float*)d_in[6];
  const float* past_ts   = (const float*)d_in[7];
  const float* ne_w1 = (const float*)d_in[8];
  const float* ne_b1 = (const float*)d_in[9];
  const float* ne_w2 = (const float*)d_in[10];
  const float* ne_b2 = (const float*)d_in[11];
  const float* emb0 = (const float*)d_in[12];
  const float* emb1 = (const float*)d_in[13];
  const float* emb2 = (const float*)d_in[14];
  const float* txt_w = (const float*)d_in[15];
  const float* txt_b = (const float*)d_in[16];
  const float* q_w = (const float*)d_in[17];
  const float* q_b = (const float*)d_in[18];
  const float* k_w = (const float*)d_in[19];
  const float* k_b = (const float*)d_in[20];
  const float* v_w = (const float*)d_in[21];
  const float* v_b = (const float*)d_in[22];
  const float* int_w = (const float*)d_in[23];
  const float* int_b = (const float*)d_in[24];
  const float* fc1_w = (const float*)d_in[25];
  const float* fc1_b = (const float*)d_in[26];
  const float* ln1_g = (const float*)d_in[27];
  const float* ln1_b = (const float*)d_in[28];
  const float* fc2_w = (const float*)d_in[29];
  const float* fc2_b = (const float*)d_in[30];
  const float* ln2_g = (const float*)d_in[31];
  const float* ln2_b = (const float*)d_in[32];
  const float* fc3_w = (const float*)d_in[33];
  const float* fc3_b = (const float*)d_in[34];
  const float* ln3_g = (const float*)d_in[35];
  const float* ln3_b = (const float*)d_in[36];
  const float* fc4_w = (const float*)d_in[37];
  const float* fc4_b = (const float*)d_in[38];
  float* out = (float*)d_out;

  // ---- workspace carve-up ----
  char* ws = (char*)d_ws;
  size_t off = 0;
  auto alloc = [&](size_t bytes) -> void* {
    off = (off + 255) & ~(size_t)255;
    void* p = ws + off;
    off += bytes;
    return p;
  };
  _Float16* txtT = (_Float16*)alloc((size_t)EMB * TXT * 2);      // [256,768]
  _Float16* qT   = (_Float16*)alloc((size_t)EMB * EMB * 2);      // [256,256]
  _Float16* kvT  = (_Float16*)alloc((size_t)512 * EMB * 2);      // [512,256]
  _Float16* intT = (_Float16*)alloc((size_t)EMB * 1024 * 2);     // [256,1024]
  _Float16* fc1T = (_Float16*)alloc((size_t)128 * 896 * 2);
  _Float16* fc2T = (_Float16*)alloc((size_t)128 * 128 * 2);
  _Float16* fc3T = (_Float16*)alloc((size_t)64 * 128 * 2);
  float*    kvb  = (float*)alloc(512 * 4);
  _Float16* pastH   = (_Float16*)alloc((size_t)BS * EMB * 2);    // past proj f16
  float*    latestF = (float*)alloc((size_t)B * EMB * 4);
  float*    qF      = (float*)alloc((size_t)B * EMB * 4);
  _Float16* kvOut   = (_Float16*)alloc((size_t)BS * 512 * 2);    // keys|vals
  _Float16* meancat = (_Float16*)alloc((size_t)B * 1024 * 2);
  _Float16* xH      = (_Float16*)alloc((size_t)B * 896 * 2);
  float*    hF      = (float*)alloc((size_t)B * 128 * 4);
  _Float16* h1H     = (_Float16*)alloc((size_t)B * 128 * 2);
  _Float16* h2H     = (_Float16*)alloc((size_t)B * 128 * 2);
  float*    h3F     = (float*)alloc((size_t)B * 64 * 4);
  float*    h3L     = (float*)alloc((size_t)B * 64 * 4);

  // ---- 1) weight prep: transpose + f16 convert ----
  dim3 tb(256);
  k_transpose_h<<<192, tb, 0, stream>>>(txt_w, txtT, TXT, EMB);
  k_transpose_h<<<64,  tb, 0, stream>>>(q_w, qT, EMB, EMB);
  k_transpose_h<<<64,  tb, 0, stream>>>(k_w, kvT, EMB, EMB);                 // rows 0..255
  k_transpose_h<<<64,  tb, 0, stream>>>(v_w, kvT + (size_t)EMB * EMB, EMB, EMB); // rows 256..511
  k_transpose_h<<<256, tb, 0, stream>>>(int_w, intT, 1024, EMB);
  k_transpose_h<<<112, tb, 0, stream>>>(fc1_w, fc1T, 896, 128);
  k_transpose_h<<<16,  tb, 0, stream>>>(fc2_w, fc2T, 128, 128);
  k_transpose_h<<<8,   tb, 0, stream>>>(fc3_w, fc3T, 128, 64);
  (void)hipMemcpyAsync(kvb, k_b, 256 * 4, hipMemcpyDeviceToDevice, stream);
  (void)hipMemcpyAsync(kvb + 256, v_b, 256 * 4, hipMemcpyDeviceToDevice, stream);

  // ---- 2) text projections (dominant GEMM) ----
  k_wmma_gemm<float><<<dim3(EMB / BN, BS / BM), tb, 0, stream>>>(
      past_text, txtT, txt_b, nullptr, pastH, (int)BS, EMB, TXT, EMB);
  k_wmma_gemm<float><<<dim3(EMB / BN, B / BM), tb, 0, stream>>>(
      cand, txtT, txt_b, latestF, nullptr, B, EMB, TXT, EMB);

  // ---- 3) q projection ----
  k_wmma_gemm<float><<<dim3(EMB / BN, B / BM), tb, 0, stream>>>(
      latestF, qT, q_b, qF, nullptr, B, EMB, EMB, EMB);

  // ---- 4) fused keys|vals projection (f16 A -> full async-to-LDS staging) ----
  k_wmma_gemm<_Float16><<<dim3(512 / BN, BS / BM), tb, 0, stream>>>(
      pastH, kvT, kvb, nullptr, kvOut, (int)BS, 512, EMB, 512);

  // ---- 5) attention + meancat (uses identical-score-row identity) ----
  k_attention<<<B / 8, tb, 0, stream>>>(qF, latestF, kvOut, past_len, past_ts,
                                        xH, meancat);

  // ---- 6) encoders + embeddings + latest into x ----
  k_encoders<<<B / 8, tb, 0, stream>>>(num_features, cat0, cat1, cat2,
                                       emb0, emb1, emb2, ne_w1, ne_b1,
                                       ne_w2, ne_b2, latestF, xH);

  // ---- 7) interaction mean (linear-of-mean identity), writes x[:,640:896) ----
  k_wmma_gemm<_Float16><<<dim3(EMB / BN, B / BM), tb, 0, stream>>>(
      meancat, intT, int_b, nullptr, xH + 640, B, EMB, 1024, 896);

  // ---- 8) final MLP ----
  k_wmma_gemm<_Float16><<<dim3(128 / BN, B / BM), tb, 0, stream>>>(
      xH, fc1T, fc1_b, hF, nullptr, B, 128, 896, 128);
  k_ln_relu<<<B / 8, tb, 0, stream>>>(hF, ln1_g, ln1_b, h1H, nullptr, 128);
  k_wmma_gemm<_Float16><<<dim3(128 / BN, B / BM), tb, 0, stream>>>(
      h1H, fc2T, fc2_b, hF, nullptr, B, 128, 128, 128);
  k_ln_relu<<<B / 8, tb, 0, stream>>>(hF, ln2_g, ln2_b, h2H, nullptr, 128);
  k_wmma_gemm<_Float16><<<dim3(64 / BN, B / BM), tb, 0, stream>>>(
      h2H, fc3T, fc3_b, h3F, nullptr, B, 64, 128, 64);
  k_ln_relu<<<B / 8, tb, 0, stream>>>(h3F, ln3_g, ln3_b, nullptr, h3L, 64);
  k_fc4<<<B / 8, tb, 0, stream>>>(h3L, fc4_w, fc4_b, out);
}
